// InvariantNet_31447750541735
// MI455X (gfx1250) — matrix operand order repaired and checked by reading, not compiled, not run on hardware
//
#include <hip/hip_runtime.h>

// ---------------- types / helpers ----------------
typedef __attribute__((ext_vector_type(16))) __bf16          v16bf;
typedef __attribute__((ext_vector_type(2)))  __bf16          v2bf;
typedef __attribute__((ext_vector_type(8)))  float           v8f;
typedef __attribute__((ext_vector_type(8)))  unsigned short  us8;
typedef __attribute__((ext_vector_type(4)))  unsigned short  us4;

#define KNN 32

static __device__ __forceinline__ unsigned short f2bf(float f) {
    unsigned u = __builtin_bit_cast(unsigned, f);
    u += 0x7fffu + ((u >> 16) & 1u);           // round-to-nearest-even
    return (unsigned short)(u >> 16);
}
// packed f32x2 -> bf16x2 (single v_cvt_pk_bf16_f32 when available)
static __device__ __forceinline__ unsigned pack2(float a, float b) {
#if __has_builtin(__builtin_amdgcn_cvt_pk_bf16_f32)
    v2bf p = __builtin_amdgcn_cvt_pk_bf16_f32(a, b);
    return __builtin_bit_cast(unsigned, p);
#else
    return (unsigned)f2bf(a) | ((unsigned)f2bf(b) << 16);
#endif
}
static __device__ __forceinline__ void wavebar() {
#if __has_builtin(__builtin_amdgcn_wave_barrier)
    __builtin_amdgcn_wave_barrier();           // scheduling fence only; DS is in-order per wave
#else
    __syncthreads();
#endif
}
static __device__ __forceinline__ __bf16 bfbits(unsigned short h) {
    return __builtin_bit_cast(__bf16, h);
}
static __device__ __forceinline__ float bf2f(unsigned short h) {
    unsigned u = ((unsigned)h) << 16;
    return __builtin_bit_cast(float, u);
}

template <int CIN> struct VecSel;
template <> struct VecSel<64>  { using T = us4; };   // CIN/16 = 4 bf16 = 8B per gather
template <> struct VecSel<128> { using T = us8; };   // CIN/16 = 8 bf16 = 16B per gather

// ---------------- KNN (top-32 by squared xy distance) + relative SE(2) coords ----------------
__global__ void knn_kernel(const float* __restrict__ pts_in, int n_in,
                           const float* __restrict__ pts_out, int n_out,
                           int* __restrict__ idx, float* __restrict__ rel) {
    __shared__ float sd[64][KNN];
    __shared__ int   si[64][KNN];
    int t = threadIdx.x;
    int m = blockIdx.x * 64 + t;
    if (m >= n_out) return;
    float ox = pts_out[m * 3 + 0], oy = pts_out[m * 3 + 1], oth = pts_out[m * 3 + 2];
    float* d  = sd[t];
    int*   id = si[t];
    float dmax = 3.4e38f; int amax = 0; int filled = 0;
    for (int j = 0; j < n_in; ++j) {
        float dx = pts_in[j * 3 + 0] - ox;
        float dy = pts_in[j * 3 + 1] - oy;
        float d2 = dx * dx + dy * dy;
        if (filled < KNN) {
            d[filled] = d2; id[filled] = j; ++filled;
            if (filled == KNN) {
                dmax = -1.f;
                for (int q = 0; q < KNN; ++q) if (d[q] > dmax) { dmax = d[q]; amax = q; }
            }
        } else if (d2 < dmax) {
            d[amax] = d2; id[amax] = j;
            dmax = -1.f;
            for (int q = 0; q < KNN; ++q) if (d[q] > dmax) { dmax = d[q]; amax = q; }
        }
    }
    float c = cosf(oth), s = sinf(oth);
    for (int k = 0; k < KNN; ++k) {
        int j = id[k];
        float dx = pts_in[j * 3 + 0] - ox;
        float dy = pts_in[j * 3 + 1] - oy;
        float rx =  c * dx + s * dy;
        float ry = -s * dx + c * dy;
        float dth = pts_in[j * 3 + 2] - oth;
        idx[m * KNN + k] = j;
        rel[(m * KNN + k) * 3 + 0] = rx;
        rel[(m * KNN + k) * 3 + 1] = ry;
        rel[(m * KNN + k) * 3 + 2] = dth;
    }
}

// ---------------- basis-MLP: phiT[m][f][k] (bf16), masked by crop radius ----------------
__global__ void phi_kernel(const float* __restrict__ rel, int n_out,
                           const float* __restrict__ w1, const float* __restrict__ b1,
                           const float* __restrict__ w2, const float* __restrict__ b2,
                           float crop2, unsigned short* __restrict__ phiT) {
    int t = blockIdx.x * blockDim.x + threadIdx.x;   // (m, k)
    if (t >= n_out * KNN) return;
    int m = t / KNN, k = t % KNN;
    float rx = rel[t * 3 + 0], ry = rel[t * 3 + 1], rz = rel[t * 3 + 2];
    float mask = (rx * rx + ry * ry <= crop2) ? 1.f : 0.f;
    float h[32];
#pragma unroll
    for (int j = 0; j < 32; ++j) {
        float a = rx * w1[j] + ry * w1[32 + j] + rz * w1[64 + j] + b1[j];
        h[j] = a > 0.f ? a : 0.f;
    }
#pragma unroll
    for (int f = 0; f < 16; ++f) {
        float p = b2[f];
#pragma unroll
        for (int j = 0; j < 32; ++j) p += h[j] * w2[j * 16 + f];
        phiT[(m * 16 + f) * 32 + k] = f2bf(p * mask);
    }
}

// ------- repack W (16,cin,cout) fp32 -> W2t (NP x Kd) bf16, TRANSPOSED, zero-padded rows ----
__global__ void w2prep_kernel(const float* __restrict__ W, int cin, int cout, int NP,
                              unsigned int* __restrict__ W2t) {   // packed pair writes
    int t = blockIdx.x * blockDim.x + threadIdx.x;
    int Kd = 16 * cin;
    int total = NP * Kd / 2;
    if (t >= total) return;
    int e0 = 2 * t;
    int o = e0 / Kd, c = e0 % Kd;
    float v0 = (o < cout) ? W[c * cout + o] : 0.f;
    float v1 = (o < cout) ? W[(c + 1) * cout + o] : 0.f;
    W2t[t] = pack2(v0, v1);
}

// ------- f32 -> chunk-interleaved bf16 mirror: vperm[row][l*(cin/16)+c] = x[row][c*16+l] ----
__global__ void cvt_perm_bf16_kernel(const float* __restrict__ x, unsigned int* __restrict__ y,
                                     size_t npairs, int cin) {
    size_t t = (size_t)blockIdx.x * blockDim.x + threadIdx.x;
    if (t >= npairs) return;
    size_t e0 = 2 * t;
    size_t row = e0 / cin;
    int p0 = (int)(e0 % cin), p1 = p0 + 1;
    int G = cin >> 4;
    const float* xr = x + row * cin;
    y[t] = pack2(xr[(p0 % G) * 16 + p0 / G], xr[(p1 % G) * 16 + p1 / G]);
}

// ---------------- t-stage: one wave per (b,m); t[r, f*cin+i] = sum_k phi[m,k,f]*v[b,idx,i] ---
// Grid must be exact: Bb*n_out waves, 8 waves/block (no early-outs: EXEC all-1 for WMMA).
template <int CIN>
__global__ void t_gather_wmma_kernel(const unsigned short* __restrict__ phiT,
                                     const int* __restrict__ idx,
                                     const unsigned short* __restrict__ vperm, // (B, n_in_pts, CIN) permuted bf16
                                     int n_out, int n_in_pts, int Bb,
                                     unsigned short* __restrict__ t) {
    using VT = typename VecSel<CIN>::T;
    constexpr int G = CIN / 16;
    __shared__ unsigned short sD[8][256];       // per-wave 16x16 bf16 D staging (no cross-wave sharing)
    int wv   = threadIdx.x >> 5;
    int lane = threadIdx.x & 31;
    int wave = (int)(blockIdx.x * 8 + wv);
    int b = wave / n_out, m = wave % n_out;
    int mu = __builtin_amdgcn_readfirstlane(m);
    int bu = __builtin_amdgcn_readfirstlane(b);
    int half = lane >> 4, l = lane & 15;

    // A operand (phi^T, 16x32): lane row f=l; elems 0..15 -> K kb..kb+15 (kb = half*16)
    v16bf a;
    {
        const unsigned short* pa = phiT + (mu * 16 + l) * 32;
        int kb = half * 16;
#pragma unroll
        for (int j = 0; j < 8; ++j) a[j] = bfbits(pa[kb + j]);
#pragma unroll
        for (int j = 0; j < 8; ++j) a[8 + j] = bfbits(pa[kb + 8 + j]);
    }
    // Neighbor rows: wave-uniform (scalar) loads; per-lane select by half.
    int rows0[16], rows1[16];
    {
        const int* pi = idx + mu * KNN;
#pragma unroll
        for (int j = 0; j < 16; ++j) {
            rows0[j] = __builtin_amdgcn_readfirstlane(pi[j]);
            rows1[j] = __builtin_amdgcn_readfirstlane(pi[16 + j]);
        }
    }
    // Wide gather: one aligned b64/b128 load per neighbor covers this lane's cols for ALL chunks.
    const unsigned short* vb = vperm + (size_t)bu * n_in_pts * CIN;
    VT gv[16];
#pragma unroll
    for (int j = 0; j < 16; ++j) {
        int row = half ? rows1[j] : rows0[j];
        gv[j] = *(const VT*)(vb + (size_t)row * CIN + l * G);
    }

    unsigned short* tr = t + (size_t)wave * 16 * CIN;
    int rr = lane >> 1, hh = lane & 1;

#pragma unroll
    for (int c = 0; c < G; ++c) {
        // B operand (32x16): lane column l; element j -> K = half*16 + j (register moves only)
        v16bf bm;
#pragma unroll
        for (int j = 0; j < 16; ++j) bm[j] = bfbits(gv[j][c]);
        v8f acc = {};
        acc = __builtin_amdgcn_wmma_f32_16x16x32_bf16(false, a, false, bm,
                                                      (short)0, acc, false, false);
        // D (lane l = tile col l, rows f=j+8*half) -> LDS -> coalesced b128 stores.
        // sD region is per-wave and DS ops are in-order per wave: compiler fence suffices.
        wavebar();
#pragma unroll
        for (int j = 0; j < 8; j += 2) {
            unsigned u = pack2(acc[j], acc[j + 1]);          // v_cvt_pk_bf16_f32
            unsigned short* s0 = &sD[wv][(j + 8 * half) * 16 + l];
            s0[0]  = (unsigned short)u;                      // row j
            s0[16] = (unsigned short)(u >> 16);              // row j+1
        }
        wavebar();
        us8 val = *(const us8*)&sD[wv][rr * 16 + hh * 8];
        *(us8*)(tr + rr * CIN + c * 16 + hh * 8) = val;
        wavebar();
    }
}

// ---------------- W-stage GEMM: C[r,o] = scale * sum_c A[r,c]*B2t[o,c]  (+ optional ReLU) ----
// B is pre-transposed (NP x Kd, L2-resident) -> pure b128 operand loads, no LDS, no barriers.
// Block = 8 waves; wave computes 16 x (16*NT); grid exact (R mult of 128, NP mult of 16*NT).
template <int NT>
__global__ void gemm2_kernel(const unsigned short* __restrict__ A,    // R x Kd bf16
                             const unsigned short* __restrict__ W2t,  // NP x Kd bf16 (transposed)
                             float* __restrict__ C,                   // R x cout f32
                             int R, int Kd, int NP, int cout, float scale, int relu) {
    int wv = threadIdx.x >> 5, lane = threadIdx.x & 31;
    int half = lane >> 4, l = lane & 15;
    int c0 = blockIdx.x * (16 * NT);
    int r0 = (blockIdx.y * 8 + wv) * 16;
    v8f acc[NT];
#pragma unroll
    for (int n = 0; n < NT; ++n) acc[n] = (v8f){};
    const unsigned short* pa0 = A + (size_t)(r0 + l) * Kd;
    for (int kk = 0; kk < Kd; kk += 32) {
        const unsigned short* pa = pa0 + kk;
        __builtin_prefetch(pa + 64, 0, 1);   // global_prefetch_b8 on streaming A
        us8 alo = *(const us8*)(pa + half * 16);
        us8 ahi = *(const us8*)(pa + half * 16 + 8);
        v16bf a;
#pragma unroll
        for (int j = 0; j < 8; ++j) { a[j] = bfbits(alo[j]); a[8 + j] = bfbits(ahi[j]); }
#pragma unroll
        for (int n = 0; n < NT; ++n) {
            const unsigned short* pb = W2t + (size_t)(c0 + n * 16 + l) * Kd + kk + half * 16;
            us8 b0 = *(const us8*)pb;
            us8 b1 = *(const us8*)(pb + 8);
            v16bf bmv;
#pragma unroll
            for (int j = 0; j < 8; ++j) { bmv[j] = bfbits(b0[j]); bmv[8 + j] = bfbits(b1[j]); }
            acc[n] = __builtin_amdgcn_wmma_f32_16x16x32_bf16(false, a, false, bmv,
                                                             (short)0, acc[n], false, false);
        }
    }
#pragma unroll
    for (int n = 0; n < NT; ++n) {
        int col = c0 + n * 16 + l;
        if (col < cout) {
#pragma unroll
            for (int j = 0; j < 8; ++j) {
                int r = r0 + j + 8 * half;
                float x = acc[n][j] * scale;
                if (relu) x = fmaxf(x, 0.f);
                C[(size_t)r * cout + col] = x;
            }
        }
    }
}

// ---------------- batch norm ----------------
__global__ void bn_stats_kernel(const float* __restrict__ x, int R, int C,
                                float* __restrict__ stats) {   // stats[c]=mean, stats[C+c]=var
    int c = blockIdx.x;
    float s = 0.f, s2 = 0.f;
    for (int r = threadIdx.x; r < R; r += blockDim.x) {
        float v = x[(size_t)r * C + c];
        s += v; s2 += v * v;
    }
    __shared__ float sh[256], sh2[256];
    sh[threadIdx.x] = s; sh2[threadIdx.x] = s2; __syncthreads();
    for (int o = 128; o > 0; o >>= 1) {
        if ((int)threadIdx.x < o) { sh[threadIdx.x] += sh[threadIdx.x + o]; sh2[threadIdx.x] += sh2[threadIdx.x + o]; }
        __syncthreads();
    }
    if (threadIdx.x == 0) {
        float m = sh[0] / (float)R;
        stats[c] = m;
        stats[C + c] = sh2[0] / (float)R - m * m;
    }
}

__global__ void bn_apply_kernel(const float* __restrict__ x, const float* __restrict__ stats,
                                const float* __restrict__ gamma, const float* __restrict__ beta,
                                float* __restrict__ y, size_t n, int C) {
    size_t t = (size_t)blockIdx.x * blockDim.x + threadIdx.x;
    if (t >= n) return;
    int c = (int)(t % C);
    float m = stats[c], v = stats[C + c];
    y[t] = (x[t] - m) * rsqrtf(v + 1e-5f) * gamma[c] + beta[c];
}

__global__ void add_kernel(float* __restrict__ y, const float* __restrict__ x, size_t n) {
    size_t t = (size_t)blockIdx.x * blockDim.x + threadIdx.x;
    if (t < n) y[t] += x[t];
}

// ---------------- first conv (cin=1) ----------------
__global__ void tfirst_kernel(const unsigned short* __restrict__ phiT, const int* __restrict__ idx,
                              const float* __restrict__ values, int n_out, int n_in, int Bb,
                              float* __restrict__ t1) {
    int t = blockIdx.x * blockDim.x + threadIdx.x;   // (b, m, f)
    int total = Bb * n_out * 16;
    if (t >= total) return;
    int f = t & 15, bm = t >> 4;
    int m = bm % n_out, b = bm / n_out;
    const unsigned short* pp = phiT + (m * 16 + f) * 32;
    const int* pi = idx + m * KNN;
    const float* vb = values + (size_t)b * n_in;
    float s = 0.f;
#pragma unroll
    for (int k = 0; k < KNN; ++k) s += bf2f(pp[k]) * vb[pi[k]];
    t1[t] = s;
}

__global__ void firstout_kernel(const float* __restrict__ t1, const float* __restrict__ Wf,
                                int n_out, int Bb, float* __restrict__ vout) {
    int t = blockIdx.x * blockDim.x + threadIdx.x;   // (b, m, o) o<64
    int total = Bb * n_out * 64;
    if (t >= total) return;
    int o = t & 63, bm = t >> 6;
    const float* tp = t1 + (size_t)bm * 16;
    float s = 0.f;
#pragma unroll
    for (int f = 0; f < 16; ++f) s += tp[f] * Wf[f * 64 + o];
    s *= (1.f / 32.f);
    vout[t] = s > 0.f ? s : 0.f;
}

// ---------------- final mean over points ----------------
__global__ void mean_kernel(const float* __restrict__ x, int N, int cout, float* __restrict__ out) {
    int b = blockIdx.x / cout, o = blockIdx.x % cout;
    float s = 0.f;
    for (int m = threadIdx.x; m < N; m += blockDim.x) s += x[((size_t)b * N + m) * cout + o];
    __shared__ float sh[256];
    sh[threadIdx.x] = s; __syncthreads();
    for (int k = 128; k > 0; k >>= 1) {
        if ((int)threadIdx.x < k) sh[threadIdx.x] += sh[threadIdx.x + k];
        __syncthreads();
    }
    if (threadIdx.x == 0) out[blockIdx.x] = sh[0] / (float)N;
}

// ================= host orchestration =================
struct ConvP { const float *w1, *b1, *w2, *b2, *W; };
struct BnP   { const float *gamma, *beta; };

extern "C" void kernel_launch(void* const* d_in, const int* in_sizes, int n_in,
                              void* d_out, int out_size, void* d_ws, size_t ws_size,
                              hipStream_t stream) {
    (void)in_sizes; (void)n_in; (void)out_size; (void)ws_size;
    const int Bb = 8, NIN = 8192, N0 = 8192, N1 = 4096, C0 = 64, C1 = 128, OUTC = 10;
    const float R0c2 = 0.04f, R1c2 = 0.08f, SCALE = 1.f / 32.f;

    const float* values = (const float*)d_in[0];   // (B, NIN, 1)
    const float* ipts   = (const float*)d_in[1];   // (NIN, 3)
    const float* pts0   = (const float*)d_in[2];   // (N0, 3)
    const float* pts1   = (const float*)d_in[3];   // (N1, 3)

    int p = 4;
    auto getConv = [&](ConvP& c) {
        c.w1 = (const float*)d_in[p++]; c.b1 = (const float*)d_in[p++];
        c.w2 = (const float*)d_in[p++]; c.b2 = (const float*)d_in[p++];
        c.W  = (const float*)d_in[p++];
    };
    auto getBn = [&](BnP& b) {
        b.gamma = (const float*)d_in[p++]; b.beta = (const float*)d_in[p++];
    };
    ConvP first; getConv(first);
    ConvP blk[2][3];
    for (int i = 0; i < 2; ++i) for (int j = 0; j < 3; ++j) getConv(blk[i][j]);
    ConvP down; getConv(down);
    BnP bns[2][4];
    for (int i = 0; i < 2; ++i) for (int j = 0; j < 4; ++j) getBn(bns[i][j]);
    BnP lastbn; getBn(lastbn);
    ConvP last; getConv(last);

    // ---- workspace bump allocator ----
    char* ws = (char*)d_ws;
    size_t off = 0;
    auto alloc = [&](size_t bytes) -> void* {
        void* q = ws + off;
        off = (off + bytes + 255) & ~(size_t)255;
        return q;
    };
    int*   idx_f = (int*)alloc((size_t)N0 * KNN * 4);
    float* rel_f = (float*)alloc((size_t)N0 * KNN * 3 * 4);
    int*   idx0  = (int*)alloc((size_t)N0 * KNN * 4);
    float* rel0  = (float*)alloc((size_t)N0 * KNN * 3 * 4);
    int*   idx1  = (int*)alloc((size_t)N1 * KNN * 4);
    float* rel1  = (float*)alloc((size_t)N1 * KNN * 3 * 4);
    int*   idx_d = (int*)alloc((size_t)N1 * KNN * 4);
    float* rel_d = (float*)alloc((size_t)N1 * KNN * 3 * 4);
    unsigned short* phiT = (unsigned short*)alloc((size_t)N0 * 16 * KNN * 2);
    unsigned short* W2   = (unsigned short*)alloc((size_t)16 * 128 * 128 * 2);
    size_t tmax = (size_t)Bb * N0 * 16 * 64 * 2;    // == Bb*N1*16*128*2
    unsigned short* tbuf = (unsigned short*)alloc(tmax);
    unsigned short* vbf  = (unsigned short*)alloc((size_t)Bb * N0 * C0 * 2);  // permuted bf16 mirror
    float* t1    = (float*)alloc((size_t)Bb * N0 * 16 * 4);
    float* vA    = (float*)alloc((size_t)Bb * N0 * C0 * 4);
    float* vB    = (float*)alloc((size_t)Bb * N0 * C0 * 4);
    float* vC    = (float*)alloc((size_t)Bb * N0 * C0 * 4);
    float* stats = (float*)alloc(256 * 4);
    float* lastout = (float*)alloc((size_t)Bb * N1 * OUTC * 4);

    auto cdiv = [](long a, long b) { return (int)((a + b - 1) / b); };

    // ---- KNN graphs ----
    knn_kernel<<<cdiv(N0, 64), 64, 0, stream>>>(ipts, NIN, pts0, N0, idx_f, rel_f);
    knn_kernel<<<cdiv(N0, 64), 64, 0, stream>>>(pts0, N0, pts0, N0, idx0, rel0);
    knn_kernel<<<cdiv(N1, 64), 64, 0, stream>>>(pts1, N1, pts1, N1, idx1, rel1);
    knn_kernel<<<cdiv(N1, 64), 64, 0, stream>>>(pts0, N0, pts1, N1, idx_d, rel_d);

    // ---- generic conv: phi -> W2t -> bf16 perm cvt -> t (gather+WMMA) -> GEMM (WMMA) ----
    auto conv = [&](const ConvP& cp, const int* idx, const float* rel, int n_out, int n_in_pts,
                    float crop2, int cin, int cout, int NP,
                    const float* vin, float* vout, int relu) {
        phi_kernel<<<cdiv((long)n_out * KNN, 256), 256, 0, stream>>>(
            rel, n_out, cp.w1, cp.b1, cp.w2, cp.b2, crop2, phiT);
        int Kd = 16 * cin;
        w2prep_kernel<<<cdiv((long)NP * Kd / 2, 256), 256, 0, stream>>>(
            cp.W, cin, cout, NP, (unsigned int*)W2);
        size_t n2 = (size_t)Bb * n_in_pts * cin / 2;
        cvt_perm_bf16_kernel<<<cdiv((long)n2, 256), 256, 0, stream>>>(
            vin, (unsigned int*)vbf, n2, cin);
        int waves = Bb * n_out;                     // multiple of 8 -> exact grid
        if (cin == 64)
            t_gather_wmma_kernel<64><<<waves / 8, 256, 0, stream>>>(
                phiT, idx, vbf, n_out, n_in_pts, Bb, tbuf);
        else
            t_gather_wmma_kernel<128><<<waves / 8, 256, 0, stream>>>(
                phiT, idx, vbf, n_out, n_in_pts, Bb, tbuf);
        int R = Bb * n_out;
        if (NP >= 64) {
            dim3 g(NP / 64, R / 128);
            gemm2_kernel<4><<<g, 256, 0, stream>>>(tbuf, W2, vout, R, Kd, NP, cout, SCALE, relu);
        } else {
            dim3 g(NP / 16, R / 128);
            gemm2_kernel<1><<<g, 256, 0, stream>>>(tbuf, W2, vout, R, Kd, NP, cout, SCALE, relu);
        }
    };
    auto bnorm = [&](const BnP& bp, const float* x, float* y, int R, int C) {
        bn_stats_kernel<<<C, 256, 0, stream>>>(x, R, C, stats);
        size_t n = (size_t)R * C;
        bn_apply_kernel<<<cdiv((long)n, 256), 256, 0, stream>>>(x, stats, bp.gamma, bp.beta, y, n, C);
    };

    // ---- first conv (cin=1): values -> vA (B,N0,64), ReLU ----
    phi_kernel<<<cdiv((long)N0 * KNN, 256), 256, 0, stream>>>(
        rel_f, N0, first.w1, first.b1, first.w2, first.b2, R0c2, phiT);
    tfirst_kernel<<<cdiv((long)Bb * N0 * 16, 256), 256, 0, stream>>>(
        phiT, idx_f, values, N0, NIN, Bb, t1);
    firstout_kernel<<<cdiv((long)Bb * N0 * 64, 256), 256, 0, stream>>>(t1, first.W, N0, Bb, vA);

    // ---- block 0 (N0, C0) ----
    {
        int R = Bb * N0;
        bnorm(bns[0][0], vA, vB, R, C0);                                   // v_in = vB
        conv(blk[0][0], idx0, rel0, N0, N0, R0c2, C0, C0, 64, vB, vC, 1);
        bnorm(bns[0][1], vC, vA, R, C0);
        conv(blk[0][1], idx0, rel0, N0, N0, R0c2, C0, C0, 64, vA, vC, 1);
        bnorm(bns[0][2], vC, vA, R, C0);
        conv(blk[0][2], idx0, rel0, N0, N0, R0c2, C0, C0, 64, vA, vC, 1);
        add_kernel<<<cdiv((long)R * C0, 256), 256, 0, stream>>>(vC, vB, (size_t)R * C0);
        bnorm(bns[0][3], vC, vA, R, C0);
        conv(down, idx_d, rel_d, N1, N0, R1c2, C0, C1, 128, vA, vB, 1);    // -> vB (B,N1,128)
    }
    // ---- block 1 (N1, C1) ----
    {
        int R = Bb * N1;
        bnorm(bns[1][0], vB, vA, R, C1);                                   // v_in = vA
        conv(blk[1][0], idx1, rel1, N1, N1, R1c2, C1, C1, 128, vA, vC, 1);
        bnorm(bns[1][1], vC, vB, R, C1);
        conv(blk[1][1], idx1, rel1, N1, N1, R1c2, C1, C1, 128, vB, vC, 1);
        bnorm(bns[1][2], vC, vB, R, C1);
        conv(blk[1][2], idx1, rel1, N1, N1, R1c2, C1, C1, 128, vB, vC, 1);
        add_kernel<<<cdiv((long)R * C1, 256), 256, 0, stream>>>(vC, vA, (size_t)R * C1);
        bnorm(lastbn, vC, vA, R, C1);
        conv(last, idx1, rel1, N1, N1, R1c2, C1, OUTC, 16, vA, lastout, 0);
    }
    // ---- mean over points -> (B, 10) ----
    mean_kernel<<<Bb * OUTC, 256, 0, stream>>>(lastout, N1, OUTC, (float*)d_out);
}
